// FullAttention_71631464563436
// MI455X (gfx1250) — matrix operand order repaired and checked by reading, not compiled, not run on hardware
//
#include <hip/hip_runtime.h>

// FlashAttention-style causal attention for gfx1250 (MI455X).
// Pass 1: pack Q,K fp32->bf16 [B,H,L,E]; Q pre-scaled by (1/sqrt(E))*log2(e)
//         so softmax runs in the exp2 domain (v_exp_f32 is natively 2^x).
//         V -> [B,H,D,S] transposed bf16.
// Pass 2: fused attention computing S^T = K*Q^T and O^T = V^T*P^T so each
//         lane owns one query row: in-register softmax reductions (v_max3 +
//         one v_permlanex16), denominator via a ones-matrix WMMA, P^T
//         assembled with v_permlanex16 (no LDS), ballot-guarded accumulator
//         rescale (skipped when the running max didn't grow), b128 stores.
// wave32; 4 waves/block; 16 query rows per wave; KV tiles of 32 keys.

typedef __attribute__((ext_vector_type(16))) __bf16 v16bf;
typedef __attribute__((ext_vector_type(8)))  __bf16 v8bf;
typedef __attribute__((ext_vector_type(8)))  float  v8f;

#define B_ 2
#define L_ 2048
#define H_ 16
#define E_ 64

union U16 { v16bf v; uint4 u[2]; int s[8]; };

__device__ __forceinline__ float fexp2(float x) {
    return __builtin_amdgcn_exp2f(x);            // v_exp_f32 (2^x)
}
__device__ __forceinline__ int permx16_i(int x) {
    // swap data between the two 16-lane halves (lane <- lane^16)
    return __builtin_amdgcn_permlanex16(x, x, 0x76543210, 0xfedcba98, false, false);
}
__device__ __forceinline__ float permx16_f(float x) {
    return __builtin_bit_cast(float, permx16_i(__builtin_bit_cast(int, x)));
}
__device__ __forceinline__ int pk_bf16(float a, float b) {
    union { __bf16 h2[2]; int u; } p;
    p.h2[0] = (__bf16)a; p.h2[1] = (__bf16)b;
    return p.u;
}

// ---------------- pack kernels ----------------
// src [B,L,H,E] f32 -> dst [B,H,L,E] bf16 (scaled); one thread per 8 elems
__global__ __launch_bounds__(256) void pack_qk_kernel(
    const float* __restrict__ src, __bf16* __restrict__ dst, float scale)
{
    int idx = blockIdx.x * blockDim.x + threadIdx.x;
    const int chunks = E_ / 8;
    int c  = idx % chunks;
    int t  = idx / chunks;
    int l  = t % L_;
    int t2 = t / L_;
    int h  = t2 % H_;
    int b  = t2 / H_;
    const float* s = src + (size_t)((b * L_ + l) * H_ + h) * E_ + c * 8;
    float4 f0 = *(const float4*)(s);
    float4 f1 = *(const float4*)(s + 4);
    v8bf o;
    o[0] = (__bf16)(f0.x * scale); o[1] = (__bf16)(f0.y * scale);
    o[2] = (__bf16)(f0.z * scale); o[3] = (__bf16)(f0.w * scale);
    o[4] = (__bf16)(f1.x * scale); o[5] = (__bf16)(f1.y * scale);
    o[6] = (__bf16)(f1.z * scale); o[7] = (__bf16)(f1.w * scale);
    *(v8bf*)(dst + (size_t)((b * H_ + h) * L_ + l) * E_ + c * 8) = o;
}

// src [B,S,H,D] f32 -> dst [B,H,D,S] bf16 (transposed); one thread per s-pair
__global__ __launch_bounds__(256) void pack_v_kernel(
    const float* __restrict__ src, __bf16* __restrict__ dst)
{
    int idx = blockIdx.x * blockDim.x + threadIdx.x;
    int d  = idx % E_;                 // d fastest: coalesced reads
    int t  = idx / E_;
    int s2 = t % (L_ / 2);
    int t2 = t / (L_ / 2);
    int h  = t2 % H_;
    int b  = t2 / H_;
    int s  = 2 * s2;
    float a0 = src[(size_t)((b * L_ + s)     * H_ + h) * E_ + d];
    float a1 = src[(size_t)((b * L_ + s + 1) * H_ + h) * E_ + d];
    *(int*)(dst + (size_t)((b * H_ + h) * E_ + d) * L_ + s) = pk_bf16(a0, a1);
}

// ---------------- attention kernel ----------------
__global__ __launch_bounds__(128) void fa_causal_wmma_kernel(
    const __bf16* __restrict__ Qbf,   // [B,H,L,E], pre-scaled (exp2 domain)
    const __bf16* __restrict__ Kbf,   // [B,H,L,E]
    const __bf16* __restrict__ Vt,    // [B,H,D,S]
    float* __restrict__ out)          // [B,L,H,D]
{
    const int lane = threadIdx.x & 31;
    const int wave = threadIdx.x >> 5;
    const int n    = lane & 15;       // this lane's query row within the tile
    const int hi   = lane >> 4;       // half-wave select
    const bool hb  = (hi != 0);

    const int nq64 = L_ / 64;
    const int tid  = blockIdx.x;
    const int qb   = tid % nq64;
    const int h    = (tid / nq64) % H_;
    const int bb   = tid / (nq64 * H_);
    const int q0   = qb * 64 + wave * 16;
    const int qrow = q0 + n;

    // Q^T B-fragments (once): lane N = query row n; elems 0..7 = e kc*32+hi*8..+7; 8..15 = +16
    const __bf16* qr = Qbf + (size_t)((bb * H_ + h) * L_ + qrow) * E_;
    U16 qB[2];
    #pragma unroll
    for (int kc = 0; kc < 2; ++kc) {
        qB[kc].u[0] = *(const uint4*)(qr + kc * 32 + hi * 8);
        qB[kc].u[1] = *(const uint4*)(qr + kc * 32 + 16 + hi * 8);
    }

    // V^T A-fragment row bases: lane M = d row within tile -> d = tt*16 + n
    const __bf16* vrow0 = Vt + (size_t)((bb * H_ + h) * E_ + n) * L_;

    // K A-fragment rows: key rows kv0 + n and kv0 + 16 + n (advanced per tile)
    const __bf16* kr0 = Kbf + (size_t)((bb * H_ + h) * L_ + n) * E_;
    const __bf16* kr1 = kr0 + (size_t)16 * E_;

    v8f o0 = {0.f,0.f,0.f,0.f,0.f,0.f,0.f,0.f};
    v8f o1 = o0, o2 = o0, o3 = o0, o4 = o0;     // o4 = softmax denominator
    float mrow = -3.0e38f;

    U16 onesA;                                   // A = all 1.0bf16 (for l)
    onesA.u[0] = make_uint4(0x3f803f80u, 0x3f803f80u, 0x3f803f80u, 0x3f803f80u);
    onesA.u[1] = onesA.u[0];

    const int ntiles = (q0 + 15) / 32 + 1;       // only the last tile is partial
    int kvoff = 0;                               // running kv0

    for (int t = 0; t < ntiles; ++t) {
        const bool masked = (t == ntiles - 1);

        // ---- S^T = K Q^T : two 16-key C tiles; lane N = query row ----
        v8f c0 = {0.f,0.f,0.f,0.f,0.f,0.f,0.f,0.f};
        v8f c1 = c0;
        #pragma unroll
        for (int kc = 0; kc < 2; ++kc) {
            U16 a0, a1;   // A elems 0..7 = e kc*32+hi*8..+7; 8..15 = +16
            a0.u[0] = *(const uint4*)(kr0 + kc * 32 + hi * 8);
            a0.u[1] = *(const uint4*)(kr0 + kc * 32 + 16 + hi * 8);
            a1.u[0] = *(const uint4*)(kr1 + kc * 32 + hi * 8);
            a1.u[1] = *(const uint4*)(kr1 + kc * 32 + 16 + hi * 8);
            c0 = __builtin_amdgcn_wmma_f32_16x16x32_bf16(false, a0.v, false, qB[kc].v,
                                                         (short)0, c0, false, false);
            c1 = __builtin_amdgcn_wmma_f32_16x16x32_bf16(false, a1.v, false, qB[kc].v,
                                                         (short)0, c1, false, false);
        }
        kr0 += 32 * E_;
        kr1 += 32 * E_;

        // ---- causal mask (last tile only); key = kvoff + r + hi*8 (+16 for c1)
        if (masked) {
            #pragma unroll
            for (int r = 0; r < 8; ++r) {
                if (kvoff + r + hi * 8      > qrow) c0[r] = -3.0e38f;
                if (kvoff + 16 + r + hi * 8 > qrow) c1[r] = -3.0e38f;
            }
        }

        // ---- per-lane row max over this lane's 16 keys + cross-half combine
        float mx = fmaxf(c0[0], c1[0]);
        #pragma unroll
        for (int r = 1; r < 8; ++r) mx = fmaxf(mx, fmaxf(c0[r], c1[r]));
        mx = fmaxf(mx, permx16_f(mx));
        const float mnew = fmaxf(mrow, mx);

        // ---- rescale accumulators only when some lane's max grew ----
        if (__ballot(mnew > mrow)) {
            const float corr = fexp2(mrow - mnew);   // scores in exp2 domain
            o0 *= corr; o1 *= corr; o2 *= corr; o3 *= corr; o4 *= corr;
        }
        mrow = mnew;

        // ---- p = exp2(s - m), packed to bf16 pairs (consecutive keys) ----
        int oc0[4], oc1[4];
        #pragma unroll
        for (int j = 0; j < 4; ++j) {
            oc0[j] = pk_bf16(fexp2(c0[2*j] - mnew), fexp2(c0[2*j+1] - mnew));
            oc1[j] = pk_bf16(fexp2(c1[2*j] - mnew), fexp2(c1[2*j+1] - mnew));
        }

        // ---- assemble P^T B-fragment: lane needs keys hi_b*16..+15 of row n.
        // lo half: [own c0 | partner c0]; hi half: [partner c1 | own c1]
        U16 pB;
        #pragma unroll
        for (int j = 0; j < 4; ++j) {
            const int z = hb ? oc0[j] : oc1[j];   // value the partner needs
            const int w = permx16_i(z);
            pB.s[j]     = hb ? w      : oc0[j];
            pB.s[4 + j] = hb ? oc1[j] : w;
        }

        // ---- O^T += V^T P^T ; l += ones * P^T ----
        const __bf16* vc = vrow0 + kvoff;
        #pragma unroll
        for (int tt = 0; tt < 4; ++tt) {
            U16 aV;
            const __bf16* vr = vc + (size_t)(tt * 16) * L_;
            aV.u[0] = *(const uint4*)(vr + hi * 16);
            aV.u[1] = *(const uint4*)(vr + hi * 16 + 8);
            v8f acc = (tt == 0) ? o0 : (tt == 1) ? o1 : (tt == 2) ? o2 : o3;
            acc = __builtin_amdgcn_wmma_f32_16x16x32_bf16(false, aV.v, false, pB.v,
                                                          (short)0, acc, false, false);
            if (tt == 0) o0 = acc; else if (tt == 1) o1 = acc;
            else if (tt == 2) o2 = acc; else o3 = acc;
        }
        o4 = __builtin_amdgcn_wmma_f32_16x16x32_bf16(false, onesA.v, false, pB.v,
                                                     (short)0, o4, false, false);
        kvoff += 32;
    }

    // ---- normalize and store: lane owns query row n; d = tt*16 + hi*8 + r ----
    const float inv = __builtin_amdgcn_rcpf(o4[0]);
    float* ob = out + (size_t)((bb * L_ + qrow) * H_ + h) * E_ + hi * 8;
    #pragma unroll
    for (int tt = 0; tt < 4; ++tt) {
        const v8f o = (tt == 0) ? o0 : (tt == 1) ? o1 : (tt == 2) ? o2 : o3;
        float4 lo4 = { o[0] * inv, o[1] * inv, o[2] * inv, o[3] * inv };
        float4 hi4 = { o[4] * inv, o[5] * inv, o[6] * inv, o[7] * inv };
        *(float4*)(ob + tt * 16)     = lo4;
        *(float4*)(ob + tt * 16 + 4) = hi4;
    }
}

extern "C" void kernel_launch(void* const* d_in, const int* in_sizes, int n_in,
                              void* d_out, int out_size, void* d_ws, size_t ws_size,
                              hipStream_t stream) {
    (void)in_sizes; (void)n_in; (void)out_size; (void)ws_size;
    const float* q = (const float*)d_in[0];
    const float* k = (const float*)d_in[1];
    const float* v = (const float*)d_in[2];
    float* out = (float*)d_out;

    const size_t elems = (size_t)B_ * H_ * L_ * E_;   // 4 Mi elements each
    __bf16* Qbf = (__bf16*)d_ws;
    __bf16* Kbf = Qbf + elems;
    __bf16* Vt  = Kbf + elems;

    const int packThreads = 256;
    const int qkBlocks = (int)(elems / 8 / packThreads);   // 2048
    const int vBlocks  = (int)(elems / 2 / packThreads);   // 8192
    // 0.125 * log2(e): run softmax in the exp2 domain (v_exp_f32 is 2^x)
    pack_qk_kernel<<<qkBlocks, packThreads, 0, stream>>>(q, Qbf, 0.18033688011112042592f);
    pack_qk_kernel<<<qkBlocks, packThreads, 0, stream>>>(k, Kbf, 1.0f);
    pack_v_kernel<<<vBlocks, packThreads, 0, stream>>>(v, Vt);

    dim3 grid(B_ * H_ * (L_ / 64));
    dim3 block(128);
    fa_causal_wmma_kernel<<<grid, block, 0, stream>>>(Qbf, Kbf, Vt, out);
}
